// QwenAttention_2173253452590
// MI455X (gfx1250) — matrix operand order repaired and checked by reading, not compiled, not run on hardware
//
#include <hip/hip_runtime.h>
#include <hip/hip_bf16.h>

// Problem constants (from reference)
#define BB 2
#define SS 2048
#define EE 2048
#define HH 16
#define KVH 4
#define GG 4
#define DD 128
#define HD (HH * DD)     // 2048
#define KVD (KVH * DD)   // 512
#define MM (BB * SS)     // 4096

typedef __attribute__((ext_vector_type(16))) __bf16 v16bf;
typedef __attribute__((ext_vector_type(8)))  float  v8f;

union AFrag { v16bf v; unsigned u[8]; };

static __device__ __forceinline__ unsigned short f2bf(float f) {
    unsigned u = __float_as_uint(f);
    unsigned r = u + 0x7FFFu + ((u >> 16) & 1u);   // round-to-nearest-even
    return (unsigned short)(r >> 16);
}

// Async global->LDS copy (CDNA5 GLOBAL_LOAD_ASYNC_TO_LDS_B128, ASYNCcnt).
// LDS operand is the low 32 bits of the flat shared-pointer (LDS aperture
// truncation per ISA 10.2); global operand is a 64-bit VGPR pair.
static __device__ __forceinline__ void async_copy_b128(void* lds_dst, const void* gsrc) {
    asm volatile("global_load_async_to_lds_b128 %0, %1, off"
                 :
                 : "v"((unsigned)(unsigned long long)lds_dst),
                   "v"((unsigned long long)gsrc)
                 : "memory");
}
static __device__ __forceinline__ void wait_async0() {
    asm volatile("s_wait_asynccnt 0x0" ::: "memory");
}

// ---------------------------------------------------------------------------
// f32 -> bf16 bulk convert (n multiple of 4)
// ---------------------------------------------------------------------------
__global__ void conv_bf16_kernel(const float* __restrict__ in,
                                 unsigned short* __restrict__ out, int n) {
    int idx = (blockIdx.x * 256 + threadIdx.x) * 4;
    if (idx + 3 < n) {
        float4 v = *(const float4*)(in + idx);
        out[idx + 0] = f2bf(v.x);
        out[idx + 1] = f2bf(v.y);
        out[idx + 2] = f2bf(v.z);
        out[idx + 3] = f2bf(v.w);
    }
}

// ---------------------------------------------------------------------------
// Generic bf16 WMMA GEMM:  C[M][N] (f32) = A[M][Kd] * Bt[N][Kd]^T (+ bias[N])
// Block tile 128x128, 256 threads = 8 waves arranged 2(M)x4(N).
// LDS double-buffered; tiles staged with async global->LDS (ASYNCcnt),
// overlapped with the 8 v_wmma of the current k-step.
// ---------------------------------------------------------------------------
__global__ void gemm_bf16_wmma(const unsigned short* __restrict__ A,
                               const unsigned short* __restrict__ Bt,
                               const float* __restrict__ bias,
                               float* __restrict__ C,
                               int M, int N, int Kd) {
    __shared__ __align__(16) unsigned short ldsA[2][128 * 32];
    __shared__ __align__(16) unsigned short ldsB[2][128 * 32];

    const int tid  = threadIdx.x;
    const int wave = tid >> 5;
    const int lane = tid & 31;
    const int l    = lane & 15;
    const int hi   = lane >> 4;
    const int wm   = (wave & 1) * 64;   // wave M offset within block tile
    const int wn   = (wave >> 1) * 32;  // wave N offset within block tile
    const int m0   = blockIdx.x * 128;
    const int n0   = blockIdx.y * 128;

    // Per-thread staging coordinates: 512 b128 chunks per tile, 2 per thread.
    const int lin0 = tid, lin1 = tid + 256;
    const int rowA0 = lin0 >> 2, kq0 = (lin0 & 3) * 8;
    const int rowA1 = lin1 >> 2, kq1 = (lin1 & 3) * 8;

    auto stage = [&](int buf, int k0) {
        async_copy_b128(&ldsA[buf][rowA0 * 32 + kq0],
                        A + (size_t)(m0 + rowA0) * Kd + k0 + kq0);
        async_copy_b128(&ldsA[buf][rowA1 * 32 + kq1],
                        A + (size_t)(m0 + rowA1) * Kd + k0 + kq1);
        async_copy_b128(&ldsB[buf][rowA0 * 32 + kq0],
                        Bt + (size_t)(n0 + rowA0) * Kd + k0 + kq0);
        async_copy_b128(&ldsB[buf][rowA1 * 32 + kq1],
                        Bt + (size_t)(n0 + rowA1) * Kd + k0 + kq1);
    };

    v8f acc[4][2];
#pragma unroll
    for (int mt = 0; mt < 4; ++mt)
#pragma unroll
        for (int nt = 0; nt < 2; ++nt)
            acc[mt][nt] = (v8f){0.f, 0.f, 0.f, 0.f, 0.f, 0.f, 0.f, 0.f};

    stage(0, 0);
    wait_async0();
    __syncthreads();

    int cur = 0;
    for (int k0 = 0; k0 < Kd; k0 += 32) {
        // Prefetch next k-step into the other buffer via the async engine.
        if (k0 + 32 < Kd) stage(cur ^ 1, k0 + 32);

        // A fragments: 16x32 bf16, ISA layout (lane<16: K 0-7,16-23; +8 hi).
        AFrag af[4];
#pragma unroll
        for (int mt = 0; mt < 4; ++mt)
#pragma unroll
            for (int i = 0; i < 8; ++i) {
                int kb = (i & 3) + ((i >> 2) << 3) + (hi << 2);   // u32 index
                af[mt].u[i] = ((const unsigned*)ldsA[cur])[(wm + mt * 16 + l) * 16 + kb];
            }
        // B fragments: 32x16 bf16 (lane = N col; same K layout as A: K 0-7,16-23; +8 hi)
        AFrag bf2[2];
#pragma unroll
        for (int nt = 0; nt < 2; ++nt)
#pragma unroll
            for (int i = 0; i < 8; ++i) {
                int kb = (i & 3) + ((i >> 2) << 3) + (hi << 2);
                bf2[nt].u[i] = ((const unsigned*)ldsB[cur])[(wn + nt * 16 + l) * 16 + kb];
            }

#pragma unroll
        for (int mt = 0; mt < 4; ++mt)
#pragma unroll
            for (int nt = 0; nt < 2; ++nt)
                acc[mt][nt] = __builtin_amdgcn_wmma_f32_16x16x32_bf16(
                    false, af[mt].v, false, bf2[nt].v,
                    (short)0, acc[mt][nt], false, false);

        wait_async0();      // our async writes to the other buffer are done
        __syncthreads();    // everyone done reading cur / writing next
        cur ^= 1;
    }

    // Epilogue: C/D layout — VGPR i holds row i (+8 for upper half), col = l.
#pragma unroll
    for (int nt = 0; nt < 2; ++nt) {
        int n = n0 + wn + nt * 16 + l;
        float bv = bias ? bias[n] : 0.f;
#pragma unroll
        for (int mt = 0; mt < 4; ++mt)
#pragma unroll
            for (int i = 0; i < 8; ++i) {
                int m = m0 + wm + mt * 16 + i + 8 * hi;
                C[(size_t)m * N + n] = acc[mt][nt][i] + bv;
            }
    }
}

// ---------------------------------------------------------------------------
// RoPE + layout transform to head-major bf16.
// q: [B*S][H*D] f32  -> [B][H][S][D] bf16   (grid: S, H, B; block: D)
// ---------------------------------------------------------------------------
__global__ void rope_q_kernel(const float* __restrict__ qf,
                              unsigned short* __restrict__ qb) {
    const int s = blockIdx.x, h = blockIdx.y, b = blockIdx.z, d = threadIdx.x;
    const float LOG_THETA = 9.210340371976184f;  // ln(10000)
    int j = d & 63;
    float invf = __expf(-((float)(2 * j) / (float)DD) * LOG_THETA);
    float fr = (float)s * invf;
    float sn, cs;
    __sincosf(fr, &sn, &cs);
    size_t idx = ((size_t)(b * SS + s)) * HD + h * DD + d;
    float t = qf[idx];
    float rot = (d < 64) ? -qf[idx + 64] : qf[idx - 64];
    qb[((size_t)(b * HH + h) * SS + s) * DD + d] = f2bf(t * cs + rot * sn);
}

__global__ void rope_k_kernel(const float* __restrict__ kf,
                              unsigned short* __restrict__ kb) {
    const int s = blockIdx.x, kv = blockIdx.y, b = blockIdx.z, d = threadIdx.x;
    const float LOG_THETA = 9.210340371976184f;
    int j = d & 63;
    float invf = __expf(-((float)(2 * j) / (float)DD) * LOG_THETA);
    float fr = (float)s * invf;
    float sn, cs;
    __sincosf(fr, &sn, &cs);
    size_t idx = ((size_t)(b * SS + s)) * KVD + kv * DD + d;
    float t = kf[idx];
    float rot = (d < 64) ? -kf[idx + 64] : kf[idx - 64];
    kb[((size_t)(b * KVH + kv) * SS + s) * DD + d] = f2bf(t * cs + rot * sn);
}

__global__ void conv_v_kernel(const float* __restrict__ vf,
                              unsigned short* __restrict__ vb) {
    const int s = blockIdx.x, kv = blockIdx.y, b = blockIdx.z, d = threadIdx.x;
    size_t idx = ((size_t)(b * SS + s)) * KVD + kv * DD + d;
    vb[((size_t)(b * KVH + kv) * SS + s) * DD + d] = f2bf(vf[idx]);
}

// ---------------------------------------------------------------------------
// Flash attention (causal, GQA). Block = 128 query rows of one head.
// 256 threads = 8 waves, each wave owns 16 query rows x full 64-key tile.
// K and V tiles both staged linearly with async global->LDS (ASYNCcnt);
// V B-fragments assembled with strided 16-bit LDS reads. Online softmax in
// f32 with half-wave shuffles (each accumulator row lives in one 16-lane
// half). Grid: (S/128, H, B), block 256. LDS = 48KB.
// ---------------------------------------------------------------------------
__global__ void flash_attn_kernel(const unsigned short* __restrict__ Qb,  // [B][H][S][D]
                                  const unsigned short* __restrict__ Kb,  // [B][KV][S][D]
                                  const unsigned short* __restrict__ Vb,  // [B][KV][S][D]
                                  unsigned short* __restrict__ Ob) {      // [B*S][H*D]
    __shared__ __align__(16) unsigned short ldsK[64 * 128];   // [key][d]
    __shared__ __align__(16) unsigned short ldsV[64 * 128];   // [key][d]
    __shared__ __align__(16) unsigned short ldsP[8 * 16 * 64];// per-wave P tile

    const int tid  = threadIdx.x;
    const int wave = tid >> 5;
    const int lane = tid & 31;
    const int l    = lane & 15;
    const int hi   = lane >> 4;
    const int qt   = blockIdx.x;
    const int h    = blockIdx.y;
    const int b    = blockIdx.z;
    const int kv   = h >> 2;          // G = 4
    const int q0   = qt * 128;

    // Load Q A-fragments for this wave's 16 rows (kept in registers).
    const size_t qbase = ((size_t)(b * HH + h) * SS + q0 + wave * 16 + l) * DD;
    AFrag qf[4];
#pragma unroll
    for (int s = 0; s < 4; ++s)
#pragma unroll
        for (int i = 0; i < 8; ++i) {
            int kb = (i & 3) * 2 + ((i >> 2) * 16) + hi * 8 + s * 32;  // element idx (even)
            qf[s].u[i] = *(const unsigned*)(Qb + qbase + kb);
        }

    v8f o[8];
#pragma unroll
    for (int td = 0; td < 8; ++td)
        o[td] = (v8f){0.f, 0.f, 0.f, 0.f, 0.f, 0.f, 0.f, 0.f};
    float mrow[8], lrow[8];
#pragma unroll
    for (int i = 0; i < 8; ++i) { mrow[i] = -3.0e38f; lrow[i] = 0.f; }

    const unsigned short* Ktb = Kb + (size_t)(b * KVH + kv) * SS * DD;
    const unsigned short* Vtb = Vb + (size_t)(b * KVH + kv) * SS * DD;
    const float SCALE = 0.08838834764831845f;  // 1/sqrt(128)

    const int ktmax = 2 * qt + 1;   // key tiles covering causal prefix
    for (int kt = 0; kt <= ktmax; ++kt) {
        __syncthreads();
        // Stage K and V tiles (16KB each, contiguous) via async global->LDS:
        // 1024 b128 chunks each, 4 per thread.
        const unsigned short* gK = Ktb + (size_t)kt * 64 * 128;
        const unsigned short* gV = Vtb + (size_t)kt * 64 * 128;
#pragma unroll
        for (int j = 0; j < 4; ++j) {
            int c = tid + j * 256;           // 0..1023
            async_copy_b128(&ldsK[c * 8], gK + c * 8);
            async_copy_b128(&ldsV[c * 8], gV + c * 8);
        }
        wait_async0();
        __syncthreads();

        // S = Q * K^T : 4 col-tiles x 4 k-steps
        v8f sc[4];
#pragma unroll
        for (int tc = 0; tc < 4; ++tc) {
            sc[tc] = (v8f){0.f, 0.f, 0.f, 0.f, 0.f, 0.f, 0.f, 0.f};
#pragma unroll
            for (int s = 0; s < 4; ++s) {
                AFrag kfr;
#pragma unroll
                for (int i = 0; i < 8; ++i) {
                    int kb = (i & 3) * 2 + ((i >> 2) * 16) + hi * 8 + 32 * s;
                    kfr.u[i] = *(const unsigned*)(ldsK + (tc * 16 + l) * 128 + kb);
                }
                sc[tc] = __builtin_amdgcn_wmma_f32_16x16x32_bf16(
                    false, qf[s].v, false, kfr.v, (short)0, sc[tc], false, false);
            }
        }

        // Scale + causal mask (only diagonal tiles actually mask).
#pragma unroll
        for (int tc = 0; tc < 4; ++tc)
#pragma unroll
            for (int i = 0; i < 8; ++i) {
                float v = sc[tc][i] * SCALE;
                int kcol = kt * 64 + tc * 16 + l;
                int qrow = q0 + wave * 16 + i + 8 * hi;
                sc[tc][i] = (kcol > qrow) ? -3.0e38f : v;
            }

        // Online softmax per row; row i+8*hi lives in VGPR i across one half.
#pragma unroll
        for (int i = 0; i < 8; ++i) {
            float tmax = fmaxf(fmaxf(sc[0][i], sc[1][i]), fmaxf(sc[2][i], sc[3][i]));
#pragma unroll
            for (int off = 8; off; off >>= 1)
                tmax = fmaxf(tmax, __shfl_xor(tmax, off, 16));
            float mnew = fmaxf(mrow[i], tmax);
            float corr = __expf(mrow[i] - mnew);
            float psum = 0.f;
#pragma unroll
            for (int tc = 0; tc < 4; ++tc) {
                float p = __expf(sc[tc][i] - mnew);
                psum += p;
                ldsP[(wave * 16 + i + 8 * hi) * 64 + tc * 16 + l] = f2bf(p);
            }
#pragma unroll
            for (int off = 8; off; off >>= 1)
                psum += __shfl_xor(psum, off, 16);
            lrow[i] = lrow[i] * corr + psum;
            mrow[i] = mnew;
#pragma unroll
            for (int td = 0; td < 8; ++td) o[td][i] *= corr;
        }

        // Wave-local LDS RAW: P writes -> P fragment reads.
        asm volatile("s_wait_dscnt 0x0" ::: "memory");

        // Reload P in A-layout (per-wave strip; no cross-wave traffic).
        AFrag pf[2];
#pragma unroll
        for (int ks = 0; ks < 2; ++ks)
#pragma unroll
            for (int i = 0; i < 8; ++i) {
                int kb = (i & 3) * 2 + ((i >> 2) * 16) + hi * 8 + ks * 32;
                pf[ks].u[i] = *(const unsigned*)(ldsP + (wave * 16 + l) * 64 + kb);
            }

        // O += P * V : 8 d-tiles x 2 k-steps.
        // V is [key][d] in LDS; B-fragment pair (k,k+1) gathered as two u16.
#pragma unroll
        for (int td = 0; td < 8; ++td)
#pragma unroll
            for (int ks = 0; ks < 2; ++ks) {
                AFrag vfr;
#pragma unroll
                for (int i = 0; i < 8; ++i) {
                    int k  = (i & 3) * 2 + ((i >> 2) * 16) + hi * 8 + 32 * ks;
                    int n  = td * 16 + l;
                    unsigned lo  = ldsV[k * 128 + n];
                    unsigned hiv = ldsV[(k + 1) * 128 + n];
                    vfr.u[i] = lo | (hiv << 16);
                }
                o[td] = __builtin_amdgcn_wmma_f32_16x16x32_bf16(
                    false, pf[ks].v, false, vfr.v, (short)0, o[td], false, false);
            }
    }

    // Normalize and store bf16 attention output as [B*S][H*D].
#pragma unroll
    for (int i = 0; i < 8; ++i) {
        float inv = 1.0f / lrow[i];
        int qg = q0 + wave * 16 + i + 8 * hi;
        size_t base = (size_t)(b * SS + qg) * HD + h * DD;
#pragma unroll
        for (int td = 0; td < 8; ++td)
            Ob[base + td * 16 + l] = f2bf(o[td][i] * inv);
    }
}

// ---------------------------------------------------------------------------
// Host-side orchestration
// ---------------------------------------------------------------------------
static inline size_t align256(size_t x) { return (x + 255) & ~(size_t)255; }

extern "C" void kernel_launch(void* const* d_in, const int* in_sizes, int n_in,
                              void* d_out, int out_size, void* d_ws, size_t ws_size,
                              hipStream_t stream) {
    (void)in_sizes; (void)n_in; (void)out_size; (void)ws_size;
    const float* x  = (const float*)d_in[0];
    const float* Wq = (const float*)d_in[1];
    const float* bq = (const float*)d_in[2];
    const float* Wk = (const float*)d_in[3];
    const float* bk = (const float*)d_in[4];
    const float* Wv = (const float*)d_in[5];
    const float* bv = (const float*)d_in[6];
    const float* Wo = (const float*)d_in[7];
    float* out = (float*)d_out;

    char* ws = (char*)d_ws;
    size_t off = 0;
    auto carve = [&](size_t bytes) { char* p = ws + off; off = align256(off + bytes); return p; };

    unsigned short* xbf   = (unsigned short*)carve((size_t)MM * EE * 2);
    unsigned short* wqbf  = (unsigned short*)carve((size_t)HD * EE * 2);
    unsigned short* wkbf  = (unsigned short*)carve((size_t)KVD * EE * 2);
    unsigned short* wvbf  = (unsigned short*)carve((size_t)KVD * EE * 2);
    unsigned short* wobf  = (unsigned short*)carve((size_t)EE * HD * 2);
    float*          qf    = (float*)carve((size_t)MM * HD * 4);
    float*          kf    = (float*)carve((size_t)MM * KVD * 4);
    float*          vf    = (float*)carve((size_t)MM * KVD * 4);
    unsigned short* qrope = (unsigned short*)carve((size_t)MM * HD * 2);
    unsigned short* krope = (unsigned short*)carve((size_t)MM * KVD * 2);
    unsigned short* vbf   = (unsigned short*)carve((size_t)MM * KVD * 2);
    // attn-out (bf16) aliases qf (f32, dead after rope_q): 16.8MB <= 33.5MB.
    unsigned short* attnbf = (unsigned short*)qf;

    // 1) bf16 conversions
    conv_bf16_kernel<<<(MM * EE) / 1024, 256, 0, stream>>>(x,  xbf,  MM * EE);
    conv_bf16_kernel<<<(HD * EE) / 1024, 256, 0, stream>>>(Wq, wqbf, HD * EE);
    conv_bf16_kernel<<<(KVD * EE) / 1024, 256, 0, stream>>>(Wk, wkbf, KVD * EE);
    conv_bf16_kernel<<<(KVD * EE) / 1024, 256, 0, stream>>>(Wv, wvbf, KVD * EE);
    conv_bf16_kernel<<<(EE * HD) / 1024, 256, 0, stream>>>(Wo, wobf, EE * HD);

    // 2) QKV projections (WMMA GEMMs, fused bias, f32 out)
    gemm_bf16_wmma<<<dim3(MM / 128, HD / 128), 256, 0, stream>>>(xbf, wqbf, bq, qf, MM, HD, EE);
    gemm_bf16_wmma<<<dim3(MM / 128, KVD / 128), 256, 0, stream>>>(xbf, wkbf, bk, kf, MM, KVD, EE);
    gemm_bf16_wmma<<<dim3(MM / 128, KVD / 128), 256, 0, stream>>>(xbf, wvbf, bv, vf, MM, KVD, EE);

    // 3) RoPE + head-major bf16 relayout
    rope_q_kernel<<<dim3(SS, HH, BB), DD, 0, stream>>>(qf, qrope);
    rope_k_kernel<<<dim3(SS, KVH, BB), DD, 0, stream>>>(kf, krope);
    conv_v_kernel<<<dim3(SS, KVH, BB), DD, 0, stream>>>(vf, vbf);

    // 4) Causal GQA flash attention (128 queries/block, 8 waves)
    flash_attn_kernel<<<dim3(SS / 128, HH, BB), 256, 0, stream>>>(qrope, krope, vbf, attnbf);

    // 5) Output projection straight into d_out (f32 [B,S,E])
    gemm_bf16_wmma<<<dim3(MM / 128, EE / 128), 256, 0, stream>>>(attnbf, wobf, nullptr, out, MM, EE, HD);
}
